// FeedForwardNetwork_12163347382319
// MI455X (gfx1250) — compile-verified, matrix operand order checked by reading, not run on hardware
//
#include <hip/hip_runtime.h>

#define NC        49      // (LMAX+1)^2 spherical-harmonic coefficients
#define NCP       64      // padded coefficient count (K for to-grid stage)
#define N_NODES   1024
#define C_IN      128
#define H_DIM     256
#define GRID_PTS  324     // 18 x 18
#define GRID_PAD  352     // 11 * 32 (zero-padded grid rows)
#define CHUNK     32      // grid rows processed per chunk
#define NCHUNK    11
#define NODE_STRIDE (H_DIM * NCP)   // 16384 bf16 elements (32KB) per node in x1t

typedef __attribute__((ext_vector_type(16))) __bf16 v16bf;
typedef __attribute__((ext_vector_type(2)))  __bf16 v2bf;
typedef __attribute__((ext_vector_type(2)))  float  v2f;
typedef __attribute__((ext_vector_type(8)))  float  v8f;

// typed pointers for the async-to-LDS builtin (param types per hipcc diagnostic)
typedef int v4i_ __attribute__((vector_size(16)));
typedef __attribute__((address_space(1))) v4i_* async_gptr;
typedef __attribute__((address_space(3))) v4i_* async_lptr;

union Frag { unsigned int u[8]; v16bf v; };

#if defined(__has_builtin)
#if __has_builtin(__builtin_amdgcn_global_load_async_to_lds_b128) && \
    __has_builtin(__builtin_amdgcn_s_wait_asynccnt)
#define USE_ASYNC_LDS 1
#endif
#endif

__constant__ int DEG_OF[NC] = {
  0,
  1,1,1,
  2,2,2,2,2,
  3,3,3,3,3,3,3,
  4,4,4,4,4,4,4,4,4,
  5,5,5,5,5,5,5,5,5,5,5,
  6,6,6,6,6,6,6,6,6,6,6,6,6};

// hardware bf16 conversions (lower to v_cvt_*bf16_f32, no software rounding)
__device__ __forceinline__ unsigned short f2bf(float x) {
  union { __bf16 b; unsigned short u; } c;
  c.b = (__bf16)x;
  return c.u;
}
__device__ __forceinline__ unsigned int pk2bf(float lo, float hi) {
  v2f f = {lo, hi};
  union { v2bf v; unsigned int u; } c;
  c.v = __builtin_convertvector(f, v2bf);
  return c.u;
}
__device__ __forceinline__ float silu_f(float x) { return x / (1.0f + __expf(-x)); }

// Load a 16x32 bf16 fragment (A-layout; B loaded identically from pre-transposed B).
// Per ISA 7.12.2: lanes 0-15 hold M=0..15 with K 0..7 / 16..23; lanes 16-31 hold K 8..15 / 24..31.
__device__ __forceinline__ v16bf frag_ld_bf16(const unsigned short* base, int row0,
                                              int ldk, int k0) {
  int lane = threadIdx.x & 31;
  int m = lane & 15, half = lane >> 4;
  const unsigned short* p = base + (size_t)(row0 + m) * (size_t)ldk + k0;
  const unsigned int* pa = (const unsigned int*)(p + half * 8);
  const unsigned int* pb = (const unsigned int*)(p + 16 + half * 8);
  Frag f;
#pragma unroll
  for (int j = 0; j < 4; ++j) { f.u[j] = pa[j]; f.u[4 + j] = pb[j]; }
  return f.v;
}

// Same fragment, sourced from f32 with on-the-fly bf16 conversion.
__device__ __forceinline__ v16bf frag_ld_f32(const float* base, int row0,
                                             size_t ldk, int k0) {
  int lane = threadIdx.x & 31;
  int m = lane & 15, half = lane >> 4;
  const float* p = base + (size_t)(row0 + m) * ldk + k0;
  Frag f;
#pragma unroll
  for (int j = 0; j < 4; ++j) {
    f.u[j]     = pk2bf(p[half * 8 + 2 * j],      p[half * 8 + 2 * j + 1]);
    f.u[4 + j] = pk2bf(p[16 + half * 8 + 2 * j], p[16 + half * 8 + 2 * j + 1]);
  }
  return f.v;
}

__device__ __forceinline__ v8f wmma_bf16(v16bf a, v16bf b, v8f c) {
  return __builtin_amdgcn_wmma_f32_16x16x32_bf16(false, a, false, b, (short)0, c,
                                                 false, false);
}

// ---------------- prep: f32 -> bf16 convert (+optional transpose, zero pad) -----
__global__ void cvt_pad(const float* __restrict__ src, unsigned short* __restrict__ dst,
                        int drows, int dcols, int srows, int scols, int do_t) {
  int idx = blockIdx.x * blockDim.x + threadIdx.x;
  if (idx >= drows * dcols) return;
  int r = idx / dcols, c = idx % dcols;
  float v = 0.f;
  if (do_t) { if (r < scols && c < srows) v = src[(size_t)c * scols + r]; }
  else      { if (r < srows && c < scols) v = src[(size_t)r * scols + c]; }
  dst[idx] = f2bf(v);
}

// ---------------- gating: silu(nf[:,0,:] @ ws + bs)  [N,H] f32 ------------------
__global__ void k_gate(const float* __restrict__ nf, const float* __restrict__ wsm,
                       const float* __restrict__ bs, float* __restrict__ gating) {
  int n = blockIdx.x, h = threadIdx.x;
  const float* x = nf + (size_t)n * NC * C_IN;
  float acc = 0.f;
#pragma unroll 4
  for (int c = 0; c < C_IN; ++c) acc = fmaf(x[c], wsm[c * H_DIM + h], acc);
  gating[n * H_DIM + h] = silu_f(acc + bs[h]);
}

// ---- SO3 linear 1: per-coef GEMM [1024,128]@[128,256], output TRANSPOSED per node:
//      x1t[node][h][coef]  (contiguous 32KB per node, async-LDS friendly)
__global__ void __launch_bounds__(256)
k_so3lin1(const float* __restrict__ nf, const unsigned short* __restrict__ w1T,
          const float* __restrict__ b1, unsigned short* __restrict__ x1t) {
  int ntile = blockIdx.x;          // node tile (16 nodes)
  int i = blockIdx.y;              // coefficient
  int w = threadIdx.x >> 5;
  int lane = threadIdx.x & 31;
  int nlane = lane & 15, half = lane >> 4;
  const float* A = nf + ((size_t)ntile * 16 * NC + i) * C_IN;
  const unsigned short* B = w1T + (size_t)DEG_OF[i] * H_DIM * C_IN;
#pragma unroll
  for (int ntl = 0; ntl < 2; ++ntl) {
    int col0 = (w * 2 + ntl) * 16;
    v8f acc = {};
#pragma unroll
    for (int ks = 0; ks < C_IN / 32; ++ks) {
      v16bf a = frag_ld_f32(A, 0, (size_t)NC * C_IN, ks * 32);
      v16bf b = frag_ld_bf16(B, col0, C_IN, ks * 32);
      acc = wmma_bf16(a, b, acc);
    }
    float bias = (i == 0) ? b1[col0 + nlane] : 0.f;
#pragma unroll
    for (int r = 0; r < 8; ++r) {
      int node = ntile * 16 + r + 8 * half;
      x1t[(size_t)node * NODE_STRIDE + (size_t)(col0 + nlane) * NCP + i] =
          f2bf(acc[r] + bias);
    }
  }
}

// one 32x256 MLP layer tile: dst = (act? silu : id)(src @ gT^T)
__device__ __forceinline__ void mlp_layer(const unsigned short* src, unsigned short* dst,
                                          const unsigned short* gT, bool act, bool storeT) {
  int w = threadIdx.x >> 5;
  int lane = threadIdx.x & 31;
  int nlane = lane & 15, half = lane >> 4;
#pragma unroll
  for (int mt = 0; mt < 2; ++mt) {
#pragma unroll
    for (int ntl = 0; ntl < 2; ++ntl) {
      int col0 = (w * 2 + ntl) * 16;
      v8f acc = {};
#pragma unroll
      for (int ks = 0; ks < H_DIM / 32; ++ks) {
        v16bf a = frag_ld_bf16(src, mt * 16, H_DIM, ks * 32);
        v16bf b = frag_ld_bf16(gT, col0, H_DIM, ks * 32);
        acc = wmma_bf16(a, b, acc);
      }
#pragma unroll
      for (int r = 0; r < 8; ++r) {
        int row = mt * 16 + r + 8 * half, col = col0 + nlane;
        float v = acc[r];
        if (act) v = silu_f(v);
        if (storeT) dst[col * CHUNK + row] = f2bf(v);      // [256][32] for B-frags
        else        dst[row * H_DIM + col] = f2bf(v);      // [32][256]
      }
    }
  }
}

// ---------------- fused per-node grid pipeline ----------------------------------
// reads  x1t[node] ([h][coef], 32KB) via async-to-LDS, writes x2 back into the
// same region in [coef][h] layout (rows 1..48 from the grid branch, row 0 = gating)
__global__ void __launch_bounds__(256)
k_grid_fused(unsigned short* __restrict__ x1t, const float* __restrict__ gating,
             const unsigned short* __restrict__ g1T, const unsigned short* __restrict__ g2T,
             const unsigned short* __restrict__ g3T, const unsigned short* __restrict__ tgb,
             const unsigned short* __restrict__ fgb) {
  __shared__ unsigned short xinT[H_DIM * NCP];    // x1[node]^T  [h][coef]  32KB
  __shared__ unsigned short bufA[CHUNK * H_DIM];  // 16KB
  __shared__ unsigned short bufB[H_DIM * CHUNK];  // 16KB (also used transposed)
  int node = blockIdx.x;
  int w = threadIdx.x >> 5;
  int lane = threadIdx.x & 31;
  int nlane = lane & 15, half = lane >> 4;

  unsigned short* nbase = x1t + (size_t)node * NODE_STRIDE;

#if defined(USE_ASYNC_LDS)
  {
    // contiguous 32KB global -> LDS via async DMA (ASYNCcnt-tracked)
    unsigned lofs = (unsigned)(unsigned long long)(&xinT[0]);
    unsigned long long g = (unsigned long long)(const char*)nbase;
#pragma unroll
    for (int it = 0; it < (H_DIM * NCP * 2) / (256 * 16); ++it) {  // 8 iters
      unsigned byteoff = (threadIdx.x + it * 256) * 16;
      __builtin_amdgcn_global_load_async_to_lds_b128(
          (async_gptr)(g + byteoff),
          (async_lptr)(unsigned)(lofs + byteoff),
          0, 0);
    }
    __builtin_amdgcn_s_wait_asynccnt(0);
  }
#else
  for (int it = threadIdx.x; it < H_DIM * NCP / 8; it += 256)
    ((uint4*)xinT)[it] = ((const uint4*)nbase)[it];
#endif
  __builtin_prefetch(g1T, 0, 1);
  __builtin_prefetch(g2T, 0, 1);
  __builtin_prefetch(g3T, 0, 1);
  __syncthreads();

  v8f acc2[4][2] = {};   // from-grid accumulators: 64 coef rows x 32 cols per wave

  for (int ch = 0; ch < NCHUNK; ++ch) {
    // stage A: bufA = to_grid[chunk] @ x1   (M=32, N=256, K=64)
#pragma unroll
    for (int mt = 0; mt < 2; ++mt) {
#pragma unroll
      for (int ntl = 0; ntl < 2; ++ntl) {
        int col0 = (w * 2 + ntl) * 16;
        v8f acc = {};
#pragma unroll
        for (int ks = 0; ks < 2; ++ks) {
          v16bf a = frag_ld_bf16(tgb, ch * CHUNK + mt * 16, NCP, ks * 32);
          v16bf b = frag_ld_bf16(xinT, col0, NCP, ks * 32);
          acc = wmma_bf16(a, b, acc);
        }
#pragma unroll
        for (int r = 0; r < 8; ++r)
          bufA[(mt * 16 + r + 8 * half) * H_DIM + col0 + nlane] = f2bf(acc[r]);
      }
    }
    __syncthreads();
    mlp_layer(bufA, bufB, g1T, true, false);   // silu(a @ g1)
    __syncthreads();
    mlp_layer(bufB, bufA, g2T, true, false);   // silu(b @ g2)
    __syncthreads();
    mlp_layer(bufA, bufB, g3T, false, true);   // (c @ g3), stored transposed [256][32]
    __syncthreads();
    // stage E: acc2 += from_grid^T[:, chunk] @ d   (M=64, N=256, K=32)
#pragma unroll
    for (int mt = 0; mt < 4; ++mt) {
#pragma unroll
      for (int ntl = 0; ntl < 2; ++ntl) {
        int col0 = (w * 2 + ntl) * 16;
        v16bf a = frag_ld_bf16(fgb, mt * 16, GRID_PAD, ch * CHUNK);
        v16bf b = frag_ld_bf16(bufB, col0, CHUNK, 0);
        acc2[mt][ntl] = wmma_bf16(a, b, acc2[mt][ntl]);
      }
    }
    __syncthreads();
  }

  // write x2 into the node's region, layout [coef][h]; rows 1..48 from grid branch
#pragma unroll
  for (int mt = 0; mt < 4; ++mt) {
#pragma unroll
    for (int ntl = 0; ntl < 2; ++ntl) {
      int col = (w * 2 + ntl) * 16 + nlane;
#pragma unroll
      for (int r = 0; r < 8; ++r) {
        int m = mt * 16 + r + 8 * half;
        if (m >= 1 && m < NC)
          nbase[m * H_DIM + col] = f2bf(acc2[mt][ntl][r]);
      }
    }
  }
  // row 0 = gated scalar branch
  nbase[threadIdx.x] = f2bf(gating[node * H_DIM + threadIdx.x]);
}

// ---------------- SO3 linear 2: per-coef GEMM [1024,256]@[256,128] -> f32 -------
// A rows = nodes, element (node,k) at x2[node*NODE_STRIDE + i*256 + k]
__global__ void __launch_bounds__(256)
k_so3lin2(const unsigned short* __restrict__ x2b, const unsigned short* __restrict__ w2T,
          const float* __restrict__ b2, float* __restrict__ out) {
  int ntile = blockIdx.x, i = blockIdx.y;
  int w = threadIdx.x >> 5;
  int lane = threadIdx.x & 31;
  int nlane = lane & 15, half = lane >> 4;
  const unsigned short* A = x2b + (size_t)ntile * 16 * NODE_STRIDE + (size_t)i * H_DIM;
  const unsigned short* B = w2T + (size_t)DEG_OF[i] * C_IN * H_DIM;
  int col0 = w * 16;
  v8f acc = {};
#pragma unroll
  for (int ks = 0; ks < H_DIM / 32; ++ks) {
    v16bf a = frag_ld_bf16(A, 0, NODE_STRIDE, ks * 32);
    v16bf b = frag_ld_bf16(B, col0, H_DIM, ks * 32);
    acc = wmma_bf16(a, b, acc);
  }
  float bias = (i == 0) ? b2[col0 + nlane] : 0.f;
#pragma unroll
  for (int r = 0; r < 8; ++r) {
    int node = ntile * 16 + r + 8 * half;
    out[((size_t)node * NC + i) * C_IN + col0 + nlane] = acc[r] + bias;
  }
}

extern "C" void kernel_launch(void* const* d_in, const int* in_sizes, int n_in,
                              void* d_out, int out_size, void* d_ws, size_t ws_size,
                              hipStream_t stream) {
  const float* nf  = (const float*)d_in[0];
  const float* w1  = (const float*)d_in[1];
  const float* b1  = (const float*)d_in[2];
  const float* w2  = (const float*)d_in[3];
  const float* b2  = (const float*)d_in[4];
  const float* wsm = (const float*)d_in[5];
  const float* bs  = (const float*)d_in[6];
  const float* g1  = (const float*)d_in[7];
  const float* g2  = (const float*)d_in[8];
  const float* g3  = (const float*)d_in[9];
  const float* tg  = (const float*)d_in[10];
  const float* fg  = (const float*)d_in[11];
  float* out = (float*)d_out;

  char* p = (char*)d_ws;
  auto alloc = [&](size_t bytes) -> char* {
    char* r = p; p += (bytes + 255) & ~(size_t)255; return r;
  };
  unsigned short* x1t = (unsigned short*)alloc((size_t)N_NODES * NODE_STRIDE * 2);
  float* gating       = (float*)alloc((size_t)N_NODES * H_DIM * 4);
  unsigned short* w1T = (unsigned short*)alloc((size_t)7 * H_DIM * C_IN * 2);
  unsigned short* w2T = (unsigned short*)alloc((size_t)7 * C_IN * H_DIM * 2);
  unsigned short* g1T = (unsigned short*)alloc((size_t)H_DIM * H_DIM * 2);
  unsigned short* g2T = (unsigned short*)alloc((size_t)H_DIM * H_DIM * 2);
  unsigned short* g3T = (unsigned short*)alloc((size_t)H_DIM * H_DIM * 2);
  unsigned short* tgb = (unsigned short*)alloc((size_t)GRID_PAD * NCP * 2);
  unsigned short* fgb = (unsigned short*)alloc((size_t)NCP * GRID_PAD * 2);

  const int T = 256;
  for (int l = 0; l < 7; ++l) {
    cvt_pad<<<(H_DIM * C_IN + T - 1) / T, T, 0, stream>>>(
        w1 + (size_t)l * C_IN * H_DIM, w1T + (size_t)l * H_DIM * C_IN,
        H_DIM, C_IN, C_IN, H_DIM, 1);
    cvt_pad<<<(C_IN * H_DIM + T - 1) / T, T, 0, stream>>>(
        w2 + (size_t)l * H_DIM * C_IN, w2T + (size_t)l * C_IN * H_DIM,
        C_IN, H_DIM, H_DIM, C_IN, 1);
  }
  cvt_pad<<<(H_DIM * H_DIM + T - 1) / T, T, 0, stream>>>(g1, g1T, H_DIM, H_DIM, H_DIM, H_DIM, 1);
  cvt_pad<<<(H_DIM * H_DIM + T - 1) / T, T, 0, stream>>>(g2, g2T, H_DIM, H_DIM, H_DIM, H_DIM, 1);
  cvt_pad<<<(H_DIM * H_DIM + T - 1) / T, T, 0, stream>>>(g3, g3T, H_DIM, H_DIM, H_DIM, H_DIM, 1);
  cvt_pad<<<(GRID_PAD * NCP + T - 1) / T, T, 0, stream>>>(tg, tgb, GRID_PAD, NCP, GRID_PTS, NC, 0);
  cvt_pad<<<(NCP * GRID_PAD + T - 1) / T, T, 0, stream>>>(fg, fgb, NCP, GRID_PAD, GRID_PTS, NC, 1);

  k_gate<<<N_NODES, H_DIM, 0, stream>>>(nf, wsm, bs, gating);
  k_so3lin1<<<dim3(N_NODES / 16, NC), 256, 0, stream>>>(nf, w1T, b1, x1t);
  k_grid_fused<<<N_NODES, 256, 0, stream>>>(x1t, gating, g1T, g2T, g3T, tgb, fgb);
  k_so3lin2<<<dim3(N_NODES / 16, NC), 256, 0, stream>>>(x1t, w2T, b2, out);
}
